// ConfidenceFilter_6390911337085
// MI455X (gfx1250) — matrix operand order repaired and checked by reading, not compiled, not run on hardware
//
#include <hip/hip_runtime.h>
#include <hip/hip_bf16.h>

// ---------------------------------------------------------------------------
// bf16-WMMA tiled GEMM:  C[M,N] = act( A[M,K] @ B[K,N] + bias[N] )
// f32 global data, f32->bf16 conversion into double-buffered LDS, f32 acc.
// Block tile 128(M) x 128(N), K-step 32, 8 waves (wave32) in a 2x4 grid,
// each wave computes a 64x32 patch = 4x2 v_wmma_f32_16x16x32_bf16 tiles.
// One barrier per K-step; global loads for step t+1 overlap WMMAs of step t.
// ---------------------------------------------------------------------------

typedef __attribute__((ext_vector_type(16))) __bf16 v16bf;
typedef __attribute__((ext_vector_type(8)))  __bf16 v8bf;
typedef __attribute__((ext_vector_type(8)))  float  v8f;
typedef __attribute__((ext_vector_type(4)))  float  v4f;

#define BM 128
#define BN 128
#define BK 32
#define SA_LD 40   // bf16 elems per LDS row (32 + 8 pad); 80 B rows, 16B aligned
#define SB_LD 40

union PkB64 { __bf16 h[4]; unsigned long long u; };
union PkB32 { __bf16 h[2]; unsigned int u; };

__global__ __launch_bounds__(256) void gemm_bf16_wmma(
    const float* __restrict__ A, const float* __restrict__ B,
    const float* __restrict__ bias, float* __restrict__ C,
    int M, int N, int K, int relu)
{
  __shared__ __attribute__((aligned(16))) __bf16 sA[2][BM * SA_LD]; // [m][k]
  __shared__ __attribute__((aligned(16))) __bf16 sB[2][BN * SB_LD]; // [n][k] (transposed)

  const int tid  = threadIdx.x;
  const int lane = tid & 31;
  const int wave = tid >> 5;      // 0..7
  const int wm   = wave >> 2;     // 0..1 : M direction (64 rows each)
  const int wn   = wave & 3;      // 0..3 : N direction (32 cols each)
  const int half = lane >> 4;     // 0/1 : half-wave
  const int l16  = lane & 15;

  const int blockM = blockIdx.x * BM;
  const int blockN = blockIdx.y * BN;

  // per-thread global->LDS work assignments (fixed across K-steps)
  // A: 128x32 f32 = 1024 float4, 4/thread
  const int aIdx0 = tid;                 // + i*256
  // B: 32x128 f32; items = 16 k-pairs x 32 n4-groups = 512, 2/thread,
  //    each item loads (2k, n4..n4+3) and (2k+1, n4..n4+3)
  const int bIdx0 = tid;                 // + i*256

  v8f acc[4][2] = {};
  v4f ra[4];        // A prefetch regs
  v4f rb[2][2];     // B prefetch regs [item][k of pair]

  const int nk = K / BK;

  // ---------------- prologue: load K-step 0 ----------------
  #pragma unroll
  for (int i = 0; i < 4; ++i) {
    int idx = aIdx0 + i * 256;
    int r   = idx >> 3;
    int c4  = (idx & 7) << 2;
    ra[i] = *(const v4f*)(A + (size_t)(blockM + r) * K + c4);
  }
  #pragma unroll
  for (int i = 0; i < 2; ++i) {
    int item = bIdx0 + i * 256;
    int kp   = item >> 5;
    int n4   = (item & 31) << 2;
    int n0   = blockN + n4;
    #pragma unroll
    for (int kk = 0; kk < 2; ++kk) {
      int k = 2 * kp + kk;
      if (n0 + 3 < N) {
        rb[i][kk] = *(const v4f*)(B + (size_t)k * N + n0);
      } else {
        const float* bp = B + (size_t)k * N;
        v4f v = {};
        if (n0 + 0 < N) v[0] = bp[n0 + 0];
        if (n0 + 1 < N) v[1] = bp[n0 + 1];
        if (n0 + 2 < N) v[2] = bp[n0 + 2];
        rb[i][kk] = v;
      }
    }
  }
  // store step 0 into buffer 0
  #pragma unroll
  for (int i = 0; i < 4; ++i) {
    int idx = aIdx0 + i * 256;
    int r   = idx >> 3;
    int c4  = (idx & 7) << 2;
    PkB64 p; p.h[0] = (__bf16)ra[i][0]; p.h[1] = (__bf16)ra[i][1];
             p.h[2] = (__bf16)ra[i][2]; p.h[3] = (__bf16)ra[i][3];
    *(unsigned long long*)&sA[0][r * SA_LD + c4] = p.u;
  }
  #pragma unroll
  for (int i = 0; i < 2; ++i) {
    int item = bIdx0 + i * 256;
    int kp   = item >> 5;
    int n4   = (item & 31) << 2;
    #pragma unroll
    for (int j = 0; j < 4; ++j) {
      PkB32 p; p.h[0] = (__bf16)rb[i][0][j]; p.h[1] = (__bf16)rb[i][1][j];
      *(unsigned int*)&sB[0][(n4 + j) * SB_LD + 2 * kp] = p.u;
    }
  }
  __syncthreads();

  // ---------------- main K loop ----------------
  for (int t = 0; t < nk; ++t) {
    const int buf  = t & 1;
    const int kt1  = (t + 1) * BK;
    const bool more = (t + 1 < nk);

    // prefetch next K-step from global (L2-resident) while we compute
    if (more) {
      #pragma unroll
      for (int i = 0; i < 4; ++i) {
        int idx = aIdx0 + i * 256;
        int r   = idx >> 3;
        int c4  = (idx & 7) << 2;
        ra[i] = *(const v4f*)(A + (size_t)(blockM + r) * K + kt1 + c4);
      }
      #pragma unroll
      for (int i = 0; i < 2; ++i) {
        int item = bIdx0 + i * 256;
        int kp   = item >> 5;
        int n4   = (item & 31) << 2;
        int n0   = blockN + n4;
        #pragma unroll
        for (int kk = 0; kk < 2; ++kk) {
          int k = kt1 + 2 * kp + kk;
          if (n0 + 3 < N) {
            rb[i][kk] = *(const v4f*)(B + (size_t)k * N + n0);
          } else {
            const float* bp = B + (size_t)k * N;
            v4f v = {};
            if (n0 + 0 < N) v[0] = bp[n0 + 0];
            if (n0 + 1 < N) v[1] = bp[n0 + 1];
            if (n0 + 2 < N) v[2] = bp[n0 + 2];
            rb[i][kk] = v;
          }
        }
      }
    }
    if (t + 2 < nk) {  // L2 prefetch two steps ahead (global_prefetch_b8)
      __builtin_prefetch(A + (size_t)(blockM + (tid >> 1)) * K + (t + 2) * BK, 0, 1);
      __builtin_prefetch(B + (size_t)((t + 2) * BK + (tid & 31)) * N + blockN, 0, 1);
    }

    // ---- fragment loads (ISA 16-bit layouts) + 8 WMMAs ----
    // A 16x32: lanes 0-15 -> K {0..7,16..23}; lanes 16-31 -> K {8..15,24..31}
    v16bf af[4], bfr[2];
    #pragma unroll
    for (int m = 0; m < 4; ++m) {
      int row = wm * 64 + m * 16 + l16;
      int kb  = half * 8;
      v8bf lo = *(const v8bf*)&sA[buf][row * SA_LD + kb];
      v8bf hi = *(const v8bf*)&sA[buf][row * SA_LD + kb + 16];
      v16bf a;
      #pragma unroll
      for (int e = 0; e < 8; ++e) { a[e] = lo[e]; a[e + 8] = hi[e]; }
      af[m] = a;
    }
    // B 32x16: lane col = L%16; lanes 0-15 -> K {0..7,16..23}; lanes 16-31 -> K {8..15,24..31}
    #pragma unroll
    for (int n = 0; n < 2; ++n) {
      int col = wn * 32 + n * 16 + l16;
      int kb  = half * 8;
      v8bf lo = *(const v8bf*)&sB[buf][col * SB_LD + kb];
      v8bf hi = *(const v8bf*)&sB[buf][col * SB_LD + kb + 16];
      v16bf b;
      #pragma unroll
      for (int e = 0; e < 8; ++e) { b[e] = lo[e]; b[e + 8] = hi[e]; }
      bfr[n] = b;
    }
    #pragma unroll
    for (int m = 0; m < 4; ++m)
      #pragma unroll
      for (int n = 0; n < 2; ++n)
        acc[m][n] = __builtin_amdgcn_wmma_f32_16x16x32_bf16(
            false, af[m], false, bfr[n], (short)0, acc[m][n], false, false);

    // ---- stage next K-step into the other buffer ----
    if (more) {
      #pragma unroll
      for (int i = 0; i < 4; ++i) {
        int idx = aIdx0 + i * 256;
        int r   = idx >> 3;
        int c4  = (idx & 7) << 2;
        PkB64 p; p.h[0] = (__bf16)ra[i][0]; p.h[1] = (__bf16)ra[i][1];
                 p.h[2] = (__bf16)ra[i][2]; p.h[3] = (__bf16)ra[i][3];
        *(unsigned long long*)&sA[buf ^ 1][r * SA_LD + c4] = p.u;
      }
      #pragma unroll
      for (int i = 0; i < 2; ++i) {
        int item = bIdx0 + i * 256;
        int kp   = item >> 5;
        int n4   = (item & 31) << 2;
        #pragma unroll
        for (int j = 0; j < 4; ++j) {
          PkB32 p; p.h[0] = (__bf16)rb[i][0][j]; p.h[1] = (__bf16)rb[i][1][j];
          *(unsigned int*)&sB[buf ^ 1][(n4 + j) * SB_LD + 2 * kp] = p.u;
        }
      }
    }
    __syncthreads();
  }

  // ---- epilogue: bias (+ReLU). D layout: VGPR r -> row r + 8*half, col L%16
  const int colBase = blockN + wn * 32;
  const bool fullN  = (colBase + 32 <= N);
  if (fullN) {
    #pragma unroll
    for (int m = 0; m < 4; ++m) {
      #pragma unroll
      for (int n = 0; n < 2; ++n) {
        int col  = colBase + n * 16 + l16;
        float bv = bias[col];
        #pragma unroll
        for (int r = 0; r < 8; ++r) {
          int row = blockM + wm * 64 + m * 16 + r + half * 8;
          float v = acc[m][n][r] + bv;
          if (relu) v = v > 0.f ? v : 0.f;
          C[(size_t)row * N + col] = v;
        }
      }
    }
  } else {
    #pragma unroll
    for (int m = 0; m < 4; ++m) {
      #pragma unroll
      for (int n = 0; n < 2; ++n) {
        int col = colBase + n * 16 + l16;
        if (col >= N) continue;
        float bv = bias[col];
        #pragma unroll
        for (int r = 0; r < 8; ++r) {
          int row = blockM + wm * 64 + m * 16 + r + half * 8;
          float v = acc[m][n][r] + bv;
          if (relu) v = v > 0.f ? v : 0.f;
          C[(size_t)row * N + col] = v;
        }
      }
    }
  }
}

// ---------------------------------------------------------------------------
// Per-row confidence + first-exit select.
// max softmax prob = 1 / sum(exp(l - max)) => confident <=> sumexp < 1/THRESH.
// d_out already holds p3; rows exiting at stage 1/2 get overwritten by p1/p2.
// ---------------------------------------------------------------------------
__global__ __launch_bounds__(256) void select_kernel(
    const float* __restrict__ p1, const float* __restrict__ p2,
    float* __restrict__ out, int C)
{
  const int row = blockIdx.x;
  const int tid = threadIdx.x;
  __shared__ float red[8];
  __shared__ float bcast;

  const float* r1 = p1 + (size_t)row * C;
  const float* r2 = p2 + (size_t)row * C;

  auto reduce_max = [&](const float* p) -> float {
    float m = -3.4e38f;
    for (int j = tid; j < C; j += 256) m = fmaxf(m, p[j]);
    for (int o = 16; o; o >>= 1) m = fmaxf(m, __shfl_xor(m, o, 32));
    if ((tid & 31) == 0) red[tid >> 5] = m;
    __syncthreads();
    if (tid == 0) {
      float mm = red[0];
      for (int w = 1; w < 8; ++w) mm = fmaxf(mm, red[w]);
      bcast = mm;
    }
    __syncthreads();
    float r = bcast;
    __syncthreads();
    return r;
  };
  auto reduce_sumexp = [&](const float* p, float m) -> float {
    float s = 0.f;
    for (int j = tid; j < C; j += 256) s += __expf(p[j] - m);
    for (int o = 16; o; o >>= 1) s += __shfl_xor(s, o, 32);
    if ((tid & 31) == 0) red[tid >> 5] = s;
    __syncthreads();
    if (tid == 0) {
      float ss = 0.f;
      for (int w = 0; w < 8; ++w) ss += red[w];
      bcast = ss;
    }
    __syncthreads();
    float r = bcast;
    __syncthreads();
    return r;
  };

  float m1 = reduce_max(r1);
  float s1 = reduce_sumexp(r1, m1);
  bool  c1 = (1.0f / s1) > 0.01f;
  float m2 = reduce_max(r2);
  float s2 = reduce_sumexp(r2, m2);
  bool  c2 = (1.0f / s2) > 0.01f;

  if (c1) {
    for (int j = tid; j < C; j += 256) out[(size_t)row * C + j] = r1[j];
  } else if (c2) {
    for (int j = tid; j < C; j += 256) out[(size_t)row * C + j] = r2[j];
  }
}

// ---------------------------------------------------------------------------
extern "C" void kernel_launch(void* const* d_in, const int* in_sizes, int n_in,
                              void* d_out, int out_size, void* d_ws, size_t ws_size,
                              hipStream_t stream) {
  const float* x   = (const float*)d_in[0];
  const float* W1  = (const float*)d_in[1];
  const float* b1  = (const float*)d_in[2];
  const float* W2  = (const float*)d_in[3];
  const float* b2  = (const float*)d_in[4];
  const float* W3  = (const float*)d_in[5];
  const float* b3  = (const float*)d_in[6];
  const float* H1w = (const float*)d_in[7];
  const float* H1b = (const float*)d_in[8];
  const float* H2w = (const float*)d_in[9];
  const float* H2b = (const float*)d_in[10];
  const float* Fw  = (const float*)d_in[11];
  const float* Fb  = (const float*)d_in[12];

  const int B = 8192, D = 2048, H = 2048, C = 1000;
  float* out = (float*)d_out;

  char* ws = (char*)d_ws;
  const size_t hBytes = (size_t)B * H * sizeof(float);   // 64 MiB
  const size_t pBytes = (size_t)B * C * sizeof(float);   // 31.25 MiB
  float* h1 = (float*)(ws);
  float* h2 = (float*)(ws + hBytes);
  float* p1 = (float*)(ws + 2 * hBytes);
  float* p2 = (float*)(ws + 2 * hBytes + pBytes);
  float* h3 = h1;  // h1 dead after stage-2 GEMMs; reuse for h3

  dim3 blk(256);
  dim3 gB(B / BM, (H + BN - 1) / BN);  // backbone: 64 x 16
  dim3 gH(B / BM, (C + BN - 1) / BN);  // heads:    64 x 8

  gemm_bf16_wmma<<<gB, blk, 0, stream>>>(x,  W1,  b1,  h1,  B, H, D, 1);
  gemm_bf16_wmma<<<gH, blk, 0, stream>>>(h1, H1w, H1b, p1,  B, C, H, 0);
  gemm_bf16_wmma<<<gB, blk, 0, stream>>>(h1, W2,  b2,  h2,  B, H, H, 1);
  gemm_bf16_wmma<<<gH, blk, 0, stream>>>(h2, H2w, H2b, p2,  B, C, H, 0);
  gemm_bf16_wmma<<<gB, blk, 0, stream>>>(h2, W3,  b3,  h3,  B, H, H, 1);
  gemm_bf16_wmma<<<gH, blk, 0, stream>>>(h3, Fw,  Fb,  out, B, C, H, 0);
  select_kernel<<<dim3(B), blk, 0, stream>>>(p1, p2, out, C);
}
